// BattagliaNMP_40484361732766
// MI455X (gfx1250) — matrix operand order repaired and checked by reading, not compiled, not run on hardware
//
#include <hip/hip_runtime.h>

typedef __attribute__((ext_vector_type(16))) __bf16 bf16x16;
typedef __attribute__((ext_vector_type(8)))  __bf16 bf16x8;
typedef __attribute__((ext_vector_type(8)))  float  f32x8;
typedef __attribute__((ext_vector_type(4)))  float  f32x4;

#define N_NODES   50000
#define N_EDGES   800000
#define HDIM      64
#define ATOM_DIM  64
#define KMSG      160      // 144 padded to multiple of 32
#define KUPD      128
#define OUT_PHIS  (N_NODES * HDIM)   // phis region offset in d_out (floats)

union AFrag { bf16x16 v; bf16x8 p[2]; };

__device__ inline bf16x8 cvt8(f32x4 x, f32x4 y) {
  bf16x8 r;
  r[0] = (__bf16)x[0]; r[1] = (__bf16)x[1]; r[2] = (__bf16)x[2]; r[3] = (__bf16)x[3];
  r[4] = (__bf16)y[0]; r[5] = (__bf16)y[1]; r[6] = (__bf16)y[2]; r[7] = (__bf16)y[3];
  return r;
}

// Async DMA one 16B chunk global -> LDS without a VGPR round trip.
// VDST operand = VGPR holding wave-relative LDS byte address (low 32 bits of
// the flat shared address per the LDS aperture mapping); GV addressing mode.
__device__ inline void async_copy_b128(const void* g, void* l) {
  unsigned laddr = (unsigned)(size_t)l;
  unsigned long long gaddr = (unsigned long long)(size_t)g;
  asm volatile("global_load_async_to_lds_b128 %0, %1, off"
               :: "v"(laddr), "v"(gaddr) : "memory");
}
__device__ inline void async_wait0() {
  asm volatile("s_wait_asynccnt 0x0" ::: "memory");
}

// ---------------------------------------------------------------------------
// Kernel 0: convert h -> bf16; build transposed, K-padded bf16 weights;
// zero d_out (h_new region is reused as the f32 agg accumulator).
// ---------------------------------------------------------------------------
__global__ __launch_bounds__(256) void prep_kernel(
    const float* __restrict__ h, const float* __restrict__ Wm,
    const float* __restrict__ Wu,
    __bf16* __restrict__ h_bf, __bf16* __restrict__ Wm_t,
    __bf16* __restrict__ Wu_t, float* __restrict__ out, int out_n)
{
  const int tid    = blockIdx.x * blockDim.x + threadIdx.x;
  const int stride = gridDim.x * blockDim.x;

  for (int i = tid; i < N_NODES * ATOM_DIM; i += stride)
    h_bf[i] = (__bf16)h[i];

  for (int i = tid; i < out_n; i += stride)
    out[i] = 0.0f;

  for (int i = tid; i < HDIM * KMSG; i += stride) {  // Wm_t[n][k], k padded
    const int n = i / KMSG, k = i % KMSG;
    Wm_t[i] = (k < 144) ? (__bf16)Wm[k * HDIM + n] : (__bf16)0.0f;
  }
  for (int i = tid; i < HDIM * KUPD; i += stride) {  // Wu_t[n][k]
    const int n = i / KUPD, k = i % KUPD;
    Wu_t[i] = (__bf16)Wu[k * HDIM + n];
  }
}

// ---------------------------------------------------------------------------
// Kernel 1: edge MLP via WMMA + atomic scatter onto agg (= d_out h_new region)
// One wave = 16 edges x 64 outputs. A built in native WMMA bf16 layout:
// lane L (<16) holds row M=L, K = kb+{0..7, 16..23}; lane L+16 holds
// row M=L, K = kb+{8..15, 24..31}.
// ---------------------------------------------------------------------------
__global__ __launch_bounds__(256) void edge_msg_kernel(
    const __bf16* __restrict__ h_bf, const float* __restrict__ e_f,
    const __bf16* __restrict__ Wm_t, const float* __restrict__ b_msg,
    const int* __restrict__ src, const int* __restrict__ dst,
    float* __restrict__ agg)
{
  __shared__ __bf16 lw[HDIM * KMSG];     // 20480 B, transposed weights
  {
    // 1280 x 16B chunks, async DMA straight into LDS (no VGPR staging)
    for (int i = threadIdx.x; i < (HDIM * KMSG * 2) / 16; i += 256)
      async_copy_b128((const char*)Wm_t + i * 16, (char*)lw + i * 16);
    async_wait0();
  }
  __syncthreads();

  const int wave = threadIdx.x >> 5;
  const int lane = threadIdx.x & 31;
  const int lo   = lane & 15;
  const int hi   = (lane >> 4) & 1;
  const int tile = blockIdx.x * 8 + wave;        // grid sized exactly: E/16/8

  const unsigned eidx = (unsigned)(tile * 16 + lo);
  const unsigned srow = (unsigned)src[eidx];
  const unsigned drow = (unsigned)dst[eidx];
  const unsigned off  = (unsigned)(hi * 8);

  AFrag a[5];
  const unsigned hsO = srow * ATOM_DIM + off;    // 32-bit elem offsets ->
  const unsigned hdO = drow * ATOM_DIM + off;    // saddr + voffset addressing
  a[0].p[0] = *(const bf16x8*)(h_bf + hsO);      // K   0..63 : h_src
  a[0].p[1] = *(const bf16x8*)(h_bf + hsO + 16);
  a[1].p[0] = *(const bf16x8*)(h_bf + hsO + 32);
  a[1].p[1] = *(const bf16x8*)(h_bf + hsO + 48);
  a[2].p[0] = *(const bf16x8*)(h_bf + hdO);      // K  64..127: h_dst
  a[2].p[1] = *(const bf16x8*)(h_bf + hdO + 16);
  a[3].p[0] = *(const bf16x8*)(h_bf + hdO + 32);
  a[3].p[1] = *(const bf16x8*)(h_bf + hdO + 48);
  {                                              // K 128..159: e (padded)
    const unsigned eO = eidx * 16u + off;
    f32x4 e0 = *(const f32x4*)(e_f + eO);
    f32x4 e1 = *(const f32x4*)(e_f + eO + 4);
    a[4].p[0] = cvt8(e0, e1);
    bf16x8 z;
#pragma unroll
    for (int i = 0; i < 8; ++i) z[i] = (__bf16)0.0f;
    a[4].p[1] = z;
  }

#pragma unroll
  for (int n = 0; n < 4; ++n) {
    const int col = n * 16 + lo;
    f32x8 acc = {0.f, 0.f, 0.f, 0.f, 0.f, 0.f, 0.f, 0.f};
#pragma unroll
    for (int kc = 0; kc < 5; ++kc) {
      // B layout: lane<16 -> col N=lo, K=kb+{0..7,16..23} ; lane>=16 -> K=kb+{8..15,24..31}
      const __bf16* wp = lw + col * KMSG + kc * 32 + hi * 8;
      AFrag b;
      b.p[0] = *(const bf16x8*)wp;
      b.p[1] = *(const bf16x8*)(wp + 16);
      acc = __builtin_amdgcn_wmma_f32_16x16x32_bf16(
          false, a[kc].v, false, b.v, (short)0, acc, false, false);
    }
    const float bias = b_msg[col];
#pragma unroll
    for (int g = 0; g < 8; ++g) {                 // D: row m = g + hi*8
      const int r = g + hi * 8;
      const unsigned dnode = (unsigned)__shfl((int)drow, r, 32);
      float v = acc[g] + bias;
      v = v > 0.f ? v : 0.f;
      __hip_atomic_fetch_add(agg + (dnode * HDIM + (unsigned)col), v,
                             __ATOMIC_RELAXED, __HIP_MEMORY_SCOPE_AGENT);
    }
  }
}

// ---------------------------------------------------------------------------
// Kernel 2: node MLP via WMMA, overwrite agg rows with h_new, atomic readout
// into phis. One wave = 16 nodes x 64 outputs, K = 64(h) + 64(agg) = 128.
// ---------------------------------------------------------------------------
__global__ __launch_bounds__(256) void node_upd_kernel(
    const __bf16* __restrict__ h_bf, const __bf16* __restrict__ Wu_t,
    const float* __restrict__ b_upd, const int* __restrict__ n2g,
    float* __restrict__ out)                       // agg in / h_new+phis out
{
  __shared__ __bf16 lw[HDIM * KUPD];               // 16384 B
  {
    for (int i = threadIdx.x; i < (HDIM * KUPD * 2) / 16; i += 256)
      async_copy_b128((const char*)Wu_t + i * 16, (char*)lw + i * 16);
    async_wait0();
  }
  __syncthreads();

  const int wave = threadIdx.x >> 5;
  const int lane = threadIdx.x & 31;
  const int lo   = lane & 15;
  const int hi   = (lane >> 4) & 1;
  const int tile = blockIdx.x * 8 + wave;
  if (tile >= N_NODES / 16) return;                // wave-uniform guard

  const unsigned node = (unsigned)(tile * 16 + lo);
  const int      gid  = n2g[node];
  const unsigned off  = (unsigned)(hi * 8);

  AFrag a[4];
  const unsigned hO = node * ATOM_DIM + off;
  a[0].p[0] = *(const bf16x8*)(h_bf + hO);         // K  0..63 : h
  a[0].p[1] = *(const bf16x8*)(h_bf + hO + 16);
  a[1].p[0] = *(const bf16x8*)(h_bf + hO + 32);
  a[1].p[1] = *(const bf16x8*)(h_bf + hO + 48);
  {                                                // K 64..127: agg (f32->bf16)
    const unsigned aO = node * HDIM + off;
    f32x4 q0 = *(const f32x4*)(out + aO);
    f32x4 q1 = *(const f32x4*)(out + aO + 4);
    f32x4 q2 = *(const f32x4*)(out + aO + 16);
    f32x4 q3 = *(const f32x4*)(out + aO + 20);
    a[2].p[0] = cvt8(q0, q1);
    a[2].p[1] = cvt8(q2, q3);
    q0 = *(const f32x4*)(out + aO + 32);
    q1 = *(const f32x4*)(out + aO + 36);
    q2 = *(const f32x4*)(out + aO + 48);
    q3 = *(const f32x4*)(out + aO + 52);
    a[3].p[0] = cvt8(q0, q1);
    a[3].p[1] = cvt8(q2, q3);
  }

#pragma unroll
  for (int n = 0; n < 4; ++n) {
    const int col = n * 16 + lo;
    f32x8 acc = {0.f, 0.f, 0.f, 0.f, 0.f, 0.f, 0.f, 0.f};
#pragma unroll
    for (int kc = 0; kc < 4; ++kc) {
      const __bf16* wp = lw + col * KUPD + kc * 32 + hi * 8;
      AFrag b;
      b.p[0] = *(const bf16x8*)wp;
      b.p[1] = *(const bf16x8*)(wp + 16);
      acc = __builtin_amdgcn_wmma_f32_16x16x32_bf16(
          false, a[kc].v, false, b.v, (short)0, acc, false, false);
    }
    const float bias = b_upd[col];
#pragma unroll
    for (int g = 0; g < 8; ++g) {
      const int r = g + hi * 8;
      float v = acc[g] + bias;
      v = v > 0.f ? v : 0.f;
      out[(unsigned)(tile * 16 + r) * HDIM + (unsigned)col] = v;   // h_new
      const unsigned gg = (unsigned)__shfl(gid, r, 32);
      __hip_atomic_fetch_add(
          out + ((unsigned)OUT_PHIS + gg * HDIM + (unsigned)col), v,
          __ATOMIC_RELAXED, __HIP_MEMORY_SCOPE_AGENT);
    }
  }
}

// ---------------------------------------------------------------------------
extern "C" void kernel_launch(void* const* d_in, const int* in_sizes, int n_in,
                              void* d_out, int out_size, void* d_ws, size_t ws_size,
                              hipStream_t stream) {
  (void)in_sizes; (void)n_in; (void)ws_size;
  const float* h     = (const float*)d_in[0];
  const float* e     = (const float*)d_in[1];
  const float* W_msg = (const float*)d_in[2];
  const float* b_msg = (const float*)d_in[3];
  const float* W_upd = (const float*)d_in[4];
  const float* b_upd = (const float*)d_in[5];
  const int*   src   = (const int*)d_in[6];
  const int*   dst   = (const int*)d_in[7];
  const int*   n2g   = (const int*)d_in[8];
  float* out = (float*)d_out;

  // workspace layout (all 16B-aligned):
  //   [0, 6.4MB)            h as bf16
  //   [+0, +20480)          W_msg transposed+padded  bf16 [64][160]
  //   [+0, +16384)          W_upd transposed         bf16 [64][128]
  char* ws = (char*)d_ws;
  __bf16* h_bf = (__bf16*)ws;
  __bf16* Wm_t = (__bf16*)(ws + (size_t)N_NODES * ATOM_DIM * 2);
  __bf16* Wu_t = (__bf16*)(ws + (size_t)N_NODES * ATOM_DIM * 2 + HDIM * KMSG * 2);

  prep_kernel<<<4096, 256, 0, stream>>>(h, W_msg, W_upd, h_bf, Wm_t, Wu_t,
                                        out, out_size);

  edge_msg_kernel<<<N_EDGES / (16 * 8), 256, 0, stream>>>(
      h_bf, e, Wm_t, b_msg, src, dst, out);

  node_upd_kernel<<<(N_NODES / 16 + 7) / 8, 256, 0, stream>>>(
      h_bf, Wu_t, b_upd, n2g, out);
}
